// Packer_71657234367155
// MI455X (gfx1250) — matrix operand order repaired and checked by reading, not compiled, not run on hardware
//
#include <hip/hip_runtime.h>
#include <hip/hip_bf16.h>
#include <math.h>

#define H       128
#define KN      48
#define LLEN    2048
#define BBATCH  2
#define NNODES  (BBATCH*LLEN)
#define C3H     384
#define EV_STRIDE 392    // halfs; 784B rows, 16B aligned
#define H1_STRIDE 136    // halfs; 272B rows, 16B aligned
#define MID_STRIDE 528   // halfs; 1056B rows, 16B aligned
#define LN_EPS  1e-5f
#define MSG_SCALE 30.0f

typedef __attribute__((ext_vector_type(16))) _Float16 v16h;
typedef __attribute__((ext_vector_type(8)))  _Float16 v8h;
typedef __attribute__((ext_vector_type(8)))  float    v8f;

// Explicit address spaces: LDS accesses lower to ds_* with 32-bit addresses (no
// generic-pointer null guards); global accesses lower to global_load/store.
#define LDS_AS __attribute__((address_space(3)))
#define GLB_AS __attribute__((address_space(1)))
typedef LDS_AS _Float16 lds_h;
typedef LDS_AS float    lds_f;
typedef GLB_AS _Float16 glb_h;
typedef GLB_AS float    glb_f;
typedef GLB_AS int      glb_i;
typedef LDS_AS v8h      lds_v8h;
typedef GLB_AS v8h      glb_v8h;
typedef GLB_AS v16h     glb_v16h;

__device__ __forceinline__ float gelu_f(float x) {
    return 0.5f * x * (1.0f + erff(x * 0.7071067811865475f));
}

__device__ __forceinline__ v8f wmma_f16(v16h a, v16h b, v8f c) {
    return __builtin_amdgcn_wmma_f32_16x16x32_f16(false, a, false, b, (short)0, c, false, false);
}

// A-fragment: 16x32 f16 tile, row-major source with `stride` halfs per row.
// lane 0-15: row M=lane, K = kbase + {0..7, 16..23}; lane 16-31: row M=lane-16, K = kbase + {8..15, 24..31}
__device__ __forceinline__ v16h load_a_lds(const lds_h* p, int stride, int mbase, int kbase) {
    int lane = threadIdx.x & 31;
    int r = mbase + (lane & 15);
    int g = lane >> 4;
    const lds_h* row = p + r * stride + kbase;
    v8h x0 = *(const lds_v8h*)(row + g * 8);
    v8h x1 = *(const lds_v8h*)(row + 16 + g * 8);
    v16h a;
#pragma unroll
    for (int i = 0; i < 8; i++) { a[i] = x0[i]; a[i + 8] = x1[i]; }
    return a;
}

__device__ __forceinline__ v16h load_a_glb(const glb_h* p, int stride, int mbase, int kbase) {
    int lane = threadIdx.x & 31;
    int r = mbase + (lane & 15);
    int g = lane >> 4;
    const glb_h* row = p + (size_t)r * stride + kbase;
    v8h x0 = *(const glb_v8h*)(row + g * 8);
    v8h x1 = *(const glb_v8h*)(row + 16 + g * 8);
    v16h a;
#pragma unroll
    for (int i = 0; i < 8; i++) { a[i] = x0[i]; a[i + 8] = x1[i]; }
    return a;
}

// B-fragment: 32x16 f16 tile from transposed weights Wt[N][Kd] (K contiguous).
// lane 0-15: col N=lane, K = kbase + {0..7, 16..23} ; lane 16-31: col N=lane-16, K = kbase + {8..15, 24..31}
__device__ __forceinline__ v16h load_b_glb(const glb_h* Wt, int Kd, int ncol0, int kbase) {
    int lane = threadIdx.x & 31;
    int c = lane & 15;
    int g = lane >> 4;
    const glb_h* col = Wt + (size_t)(ncol0 + c) * Kd + kbase;
    v8h x0 = *(const glb_v8h*)(col + g * 8);
    v8h x1 = *(const glb_v8h*)(col + 16 + g * 8);
    v16h b;
#pragma unroll
    for (int i = 0; i < 8; i++) { b[i] = x0[i]; b[i + 8] = x1[i]; }
    return b;
}

__device__ __forceinline__ float block_sum128(float v, lds_f* sRed) {
    int t = threadIdx.x;
    sRed[t] = v;
    __syncthreads();
#pragma unroll
    for (int off = 64; off > 0; off >>= 1) {
        if (t < off) sRed[t] += sRed[t + off];
        __syncthreads();
    }
    float r = sRed[0];
    __syncthreads();
    return r;
}

// 3-layer MLP on a 48x384 tile: [48x384]@[384x128] gelu -> @[128x128] gelu -> @[128x128]
// Result (optionally row-masked, +b3) written to sMsg[48][128] f32. Block = 128 threads (4 waves).
__device__ __forceinline__ void mlp3_msg(
        const lds_h* sEV, lds_h* sH1, lds_h* sH2, lds_f* sMsg,
        const glb_h* W1t, const glb_f* b1,
        const glb_h* W2t, const glb_f* b2,
        const glb_h* W3t, const glb_f* b3,
        const glb_f* maskRow) {
    const int wave = threadIdx.x >> 5;
    const int lane = threadIdx.x & 31;
    const int c = lane & 15, g = lane >> 4;
    const int n0 = wave * 32;

    v8f acc[3][2];
    // ---- layer 1: K = 384 ----
#pragma unroll
    for (int mt = 0; mt < 3; mt++)
#pragma unroll
        for (int nt = 0; nt < 2; nt++)
#pragma unroll
            for (int i = 0; i < 8; i++) acc[mt][nt][i] = 0.0f;
#pragma unroll
    for (int kt = 0; kt < 12; kt++) {
        v16h b0 = load_b_glb(W1t, C3H, n0, kt * 32);
        v16h b1f = load_b_glb(W1t, C3H, n0 + 16, kt * 32);
#pragma unroll
        for (int mt = 0; mt < 3; mt++) {
            v16h a = load_a_lds(sEV, EV_STRIDE, mt * 16, kt * 32);
            acc[mt][0] = wmma_f16(a, b0, acc[mt][0]);
            acc[mt][1] = wmma_f16(a, b1f, acc[mt][1]);
        }
    }
#pragma unroll
    for (int mt = 0; mt < 3; mt++)
#pragma unroll
        for (int nt = 0; nt < 2; nt++) {
            int col = n0 + nt * 16 + c;
            float bias = b1[col];
#pragma unroll
            for (int v = 0; v < 8; v++) {
                int row = mt * 16 + g * 8 + v;
                sH1[row * H1_STRIDE + col] = (_Float16)gelu_f(acc[mt][nt][v] + bias);
            }
        }
    __syncthreads();

    // ---- layer 2: K = 128 ----
#pragma unroll
    for (int mt = 0; mt < 3; mt++)
#pragma unroll
        for (int nt = 0; nt < 2; nt++)
#pragma unroll
            for (int i = 0; i < 8; i++) acc[mt][nt][i] = 0.0f;
#pragma unroll
    for (int kt = 0; kt < 4; kt++) {
        v16h b0 = load_b_glb(W2t, H, n0, kt * 32);
        v16h b1f = load_b_glb(W2t, H, n0 + 16, kt * 32);
#pragma unroll
        for (int mt = 0; mt < 3; mt++) {
            v16h a = load_a_lds(sH1, H1_STRIDE, mt * 16, kt * 32);
            acc[mt][0] = wmma_f16(a, b0, acc[mt][0]);
            acc[mt][1] = wmma_f16(a, b1f, acc[mt][1]);
        }
    }
#pragma unroll
    for (int mt = 0; mt < 3; mt++)
#pragma unroll
        for (int nt = 0; nt < 2; nt++) {
            int col = n0 + nt * 16 + c;
            float bias = b2[col];
#pragma unroll
            for (int v = 0; v < 8; v++) {
                int row = mt * 16 + g * 8 + v;
                sH2[row * H1_STRIDE + col] = (_Float16)gelu_f(acc[mt][nt][v] + bias);
            }
        }
    __syncthreads();

    // ---- layer 3: K = 128, f32 out (sMsg aliases sEV; sEV no longer read) ----
#pragma unroll
    for (int mt = 0; mt < 3; mt++)
#pragma unroll
        for (int nt = 0; nt < 2; nt++)
#pragma unroll
            for (int i = 0; i < 8; i++) acc[mt][nt][i] = 0.0f;
#pragma unroll
    for (int kt = 0; kt < 4; kt++) {
        v16h b0 = load_b_glb(W3t, H, n0, kt * 32);
        v16h b1f = load_b_glb(W3t, H, n0 + 16, kt * 32);
#pragma unroll
        for (int mt = 0; mt < 3; mt++) {
            v16h a = load_a_lds(sH2, H1_STRIDE, mt * 16, kt * 32);
            acc[mt][0] = wmma_f16(a, b0, acc[mt][0]);
            acc[mt][1] = wmma_f16(a, b1f, acc[mt][1]);
        }
    }
#pragma unroll
    for (int mt = 0; mt < 3; mt++)
#pragma unroll
        for (int nt = 0; nt < 2; nt++) {
            int col = n0 + nt * 16 + c;
            float bias = b3[col];
#pragma unroll
            for (int v = 0; v < 8; v++) {
                int row = mt * 16 + g * 8 + v;
                float val = acc[mt][nt][v] + bias;
                if (maskRow) val *= maskRow[row];
                sMsg[row * H + col] = val;
            }
        }
    __syncthreads();
}

// ---------------- weight transpose + fp32 -> fp16 ----------------
__global__ void convert_tr_kernel(const float* src, _Float16* dst, int Kd, int Nd) {
    const glb_f* s = (const glb_f*)src;
    glb_h* d = (glb_h*)dst;
    int i = blockIdx.x * 256 + threadIdx.x;
    if (i < Kd * Nd) {
        int k = i / Nd, n = i % Nd;
        d[(size_t)n * Kd + k] = (_Float16)s[(size_t)k * Nd + n];
    }
}

// ---------------- node message + LN1 ----------------
__global__ void __launch_bounds__(128) node_msg_kernel(
    const float* hV_, const float* hE_, const int* Eidx_, const float* maskA_,
    const _Float16* W1t_, const float* b1_,
    const _Float16* W2t_, const float* b2_,
    const _Float16* W3t_, const float* b3_,
    const float* g1_, const float* be1_,
    float* hln_f32_, _Float16* hln_h_) {
    __shared__ _Float16 sEVs[KN * EV_STRIDE];
    __shared__ _Float16 sH1s[KN * H1_STRIDE];
    __shared__ _Float16 sH2s[KN * H1_STRIDE];
    __shared__ float sReds[128];
    lds_h* sEV = (lds_h*)sEVs;
    lds_h* sH1 = (lds_h*)sH1s;
    lds_h* sH2 = (lds_h*)sH2s;
    lds_f* sRed = (lds_f*)sReds;
    lds_f* sMsg = (lds_f*)sEVs;  // aliases sEV after layer 1

    const glb_f* hV = (const glb_f*)hV_;
    const glb_f* hE = (const glb_f*)hE_;
    const glb_i* Eidx = (const glb_i*)Eidx_;
    const glb_f* maskA = (const glb_f*)maskA_;

    const int node = blockIdx.x;
    const int t = threadIdx.x;
    const int bq = node / LLEN;
    const glb_i* eidx = Eidx + (size_t)node * KN;

    float hv_c = hV[(size_t)node * H + t];
    for (int k = 0; k < KN; k++) {
        int nb = eidx[k];
        sEV[k * EV_STRIDE + t]         = (_Float16)hv_c;
        sEV[k * EV_STRIDE + H + t]     = (_Float16)hE[((size_t)node * KN + k) * H + t];
        sEV[k * EV_STRIDE + 2 * H + t] = (_Float16)hV[((size_t)(bq * LLEN + nb)) * H + t];
    }
    __syncthreads();

    mlp3_msg(sEV, sH1, sH2, sMsg,
             (const glb_h*)W1t_, (const glb_f*)b1_,
             (const glb_h*)W2t_, (const glb_f*)b2_,
             (const glb_h*)W3t_, (const glb_f*)b3_,
             maskA + (size_t)node * KN);

    float s = 0.0f;
    for (int k = 0; k < KN; k++) s += sMsg[k * H + t];
    float x = hv_c + s * (1.0f / MSG_SCALE);
    float mu = block_sum128(x, sRed) * (1.0f / H);
    float d = x - mu;
    float var = block_sum128(d * d, sRed) * (1.0f / H);
    float y = d * rsqrtf(var + LN_EPS) * ((const glb_f*)g1_)[t] + ((const glb_f*)be1_)[t];
    ((glb_f*)hln_f32_)[(size_t)node * H + t] = y;
    ((glb_h*)hln_h_)[(size_t)node * H + t] = (_Float16)y;
}

// ---------------- FFN + LN2 + mask_V ----------------
__global__ void __launch_bounds__(128) ffn_kernel(
    const _Float16* hln_h_, const float* hln_f32_,
    const _Float16* Wint_, const float* binp_,
    const _Float16* Woutt_, const float* boutp_,
    const float* g2_, const float* be2_,
    const float* maskV_, float* outV_) {
    __shared__ _Float16 sMids[16 * MID_STRIDE];
    __shared__ float sOuts[16 * H];
    lds_h* sMid = (lds_h*)sMids;
    lds_f* sOut = (lds_f*)sOuts;

    const glb_h* hln_h = (const glb_h*)hln_h_;
    const glb_f* hln_f32 = (const glb_f*)hln_f32_;
    const glb_h* Wint = (const glb_h*)Wint_;
    const glb_h* Woutt = (const glb_h*)Woutt_;
    const glb_f* binp = (const glb_f*)binp_;
    const glb_f* boutp = (const glb_f*)boutp_;
    const glb_f* g2 = (const glb_f*)g2_;
    const glb_f* be2 = (const glb_f*)be2_;
    const glb_f* maskV = (const glb_f*)maskV_;
    glb_f* outV = (glb_f*)outV_;

    const int m0 = blockIdx.x * 16;
    const int wave = threadIdx.x >> 5, lane = threadIdx.x & 31;
    const int c = lane & 15, g = lane >> 4;

    // GEMM1: [16x128] @ [128x512], each wave owns 8 N-tiles
    v8f acc[8];
#pragma unroll
    for (int j = 0; j < 8; j++)
#pragma unroll
        for (int i = 0; i < 8; i++) acc[j][i] = 0.0f;
    const glb_h* Abase = hln_h + (size_t)m0 * H;
#pragma unroll
    for (int kt = 0; kt < 4; kt++) {
        v16h a = load_a_glb(Abase, H, 0, kt * 32);
#pragma unroll
        for (int j = 0; j < 8; j++) {
            v16h b = load_b_glb(Wint, H, wave * 128 + j * 16, kt * 32);
            acc[j] = wmma_f16(a, b, acc[j]);
        }
    }
#pragma unroll
    for (int j = 0; j < 8; j++) {
        int col = wave * 128 + j * 16 + c;
        float bias = binp[col];
#pragma unroll
        for (int v = 0; v < 8; v++) {
            int row = g * 8 + v;
            sMid[row * MID_STRIDE + col] = (_Float16)gelu_f(acc[j][v] + bias);
        }
    }
    __syncthreads();

    // GEMM2: [16x512] @ [512x128], each wave owns 2 N-tiles
    v8f acc2[2];
#pragma unroll
    for (int nt = 0; nt < 2; nt++)
#pragma unroll
        for (int i = 0; i < 8; i++) acc2[nt][i] = 0.0f;
#pragma unroll
    for (int kt = 0; kt < 16; kt++) {
        v16h a = load_a_lds(sMid, MID_STRIDE, 0, kt * 32);
        acc2[0] = wmma_f16(a, load_b_glb(Woutt, 512, wave * 32, kt * 32), acc2[0]);
        acc2[1] = wmma_f16(a, load_b_glb(Woutt, 512, wave * 32 + 16, kt * 32), acc2[1]);
    }
#pragma unroll
    for (int nt = 0; nt < 2; nt++) {
        int col = wave * 32 + nt * 16 + c;
        float bias = boutp[col];
#pragma unroll
        for (int v = 0; v < 8; v++) {
            int row = g * 8 + v;
            sOut[row * H + col] = acc2[nt][v] + bias + hln_f32[((size_t)(m0 + row)) * H + col];
        }
    }
    __syncthreads();

    // per-node LN over 128 + mask_V; one wave per 4 rows, lane owns 4 cols
    for (int r = wave; r < 16; r += 4) {
        float xv[4], s = 0.0f, s2 = 0.0f;
#pragma unroll
        for (int q = 0; q < 4; q++) {
            xv[q] = sOut[r * H + lane * 4 + q];
            s += xv[q]; s2 += xv[q] * xv[q];
        }
#pragma unroll
        for (int off = 16; off > 0; off >>= 1) {
            s += __shfl_xor(s, off); s2 += __shfl_xor(s2, off);
        }
        float mu = s * (1.0f / H);
        float var = s2 * (1.0f / H) - mu * mu;
        float inv = rsqrtf(var + LN_EPS);
        int nodeq = m0 + r;
        float mv = maskV[nodeq];
#pragma unroll
        for (int q = 0; q < 4; q++) {
            int col = lane * 4 + q;
            outV[(size_t)nodeq * H + col] = mv * ((xv[q] - mu) * inv * g2[col] + be2[col]);
        }
    }
}

// ---------------- edge update + LN3 ----------------
__global__ void __launch_bounds__(128) edge_kernel(
    const float* hVnew_, const float* hE_in_, const int* Eidx_,
    const _Float16* W11t_, const float* b11_,
    const _Float16* W12t_, const float* b12_,
    const _Float16* W13t_, const float* b13_,
    const float* g3_, const float* be3_,
    float* outE_) {
    __shared__ _Float16 sEVs[KN * EV_STRIDE];
    __shared__ _Float16 sH1s[KN * H1_STRIDE];
    __shared__ _Float16 sH2s[KN * H1_STRIDE];
    lds_h* sEV = (lds_h*)sEVs;
    lds_h* sH1 = (lds_h*)sH1s;
    lds_h* sH2 = (lds_h*)sH2s;
    lds_f* sMsg = (lds_f*)sEVs;

    const glb_f* hVnew = (const glb_f*)hVnew_;
    const glb_f* hE_in = (const glb_f*)hE_in_;
    const glb_i* Eidx = (const glb_i*)Eidx_;
    const glb_f* g3 = (const glb_f*)g3_;
    const glb_f* be3 = (const glb_f*)be3_;
    glb_f* outE = (glb_f*)outE_;

    const int node = blockIdx.x;
    const int t = threadIdx.x;
    const int bq = node / LLEN;
    const glb_i* eidx = Eidx + (size_t)node * KN;

    float hv_c = hVnew[(size_t)node * H + t];
    for (int k = 0; k < KN; k++) {
        int nb = eidx[k];
        sEV[k * EV_STRIDE + t]         = (_Float16)hv_c;
        sEV[k * EV_STRIDE + H + t]     = (_Float16)hE_in[((size_t)node * KN + k) * H + t];
        sEV[k * EV_STRIDE + 2 * H + t] = (_Float16)hVnew[((size_t)(bq * LLEN + nb)) * H + t];
    }
    __syncthreads();

    mlp3_msg(sEV, sH1, sH2, sMsg,
             (const glb_h*)W11t_, (const glb_f*)b11_,
             (const glb_h*)W12t_, (const glb_f*)b12_,
             (const glb_h*)W13t_, (const glb_f*)b13_,
             nullptr);

    // per-edge-row LN(h_E + msg); wave handles 12 rows, lane owns 4 cols
    const int wave = threadIdx.x >> 5, lane = threadIdx.x & 31;
    for (int r = wave; r < KN; r += 4) {
        float xv[4], s = 0.0f, s2 = 0.0f;
#pragma unroll
        for (int q = 0; q < 4; q++) {
            int col = lane * 4 + q;
            xv[q] = hE_in[((size_t)node * KN + r) * H + col] + sMsg[r * H + col];
            s += xv[q]; s2 += xv[q] * xv[q];
        }
#pragma unroll
        for (int off = 16; off > 0; off >>= 1) {
            s += __shfl_xor(s, off); s2 += __shfl_xor(s2, off);
        }
        float mu = s * (1.0f / H);
        float var = s2 * (1.0f / H) - mu * mu;
        float inv = rsqrtf(var + LN_EPS);
#pragma unroll
        for (int q = 0; q < 4; q++) {
            int col = lane * 4 + q;
            outE[((size_t)node * KN + r) * H + col] = (xv[q] - mu) * inv * g3[col] + be3[col];
        }
    }
}

extern "C" void kernel_launch(void* const* d_in, const int* in_sizes, int n_in,
                              void* d_out, int out_size, void* d_ws, size_t ws_size,
                              hipStream_t stream) {
    (void)in_sizes; (void)n_in; (void)out_size; (void)ws_size;
    const float* hV    = (const float*)d_in[0];
    const float* hE    = (const float*)d_in[1];
    const int*   Eidx  = (const int*)d_in[2];
    const float* maskV = (const float*)d_in[3];
    const float* maskA = (const float*)d_in[4];
    const float* W1  = (const float*)d_in[5];  const float* b1  = (const float*)d_in[6];
    const float* W2  = (const float*)d_in[7];  const float* b2  = (const float*)d_in[8];
    const float* W3  = (const float*)d_in[9];  const float* b3  = (const float*)d_in[10];
    const float* W11 = (const float*)d_in[11]; const float* b11 = (const float*)d_in[12];
    const float* W12 = (const float*)d_in[13]; const float* b12 = (const float*)d_in[14];
    const float* W13 = (const float*)d_in[15]; const float* b13 = (const float*)d_in[16];
    const float* Win  = (const float*)d_in[17]; const float* binp  = (const float*)d_in[18];
    const float* Wout = (const float*)d_in[19]; const float* boutp = (const float*)d_in[20];
    const float* g1 = (const float*)d_in[21]; const float* be1 = (const float*)d_in[22];
    const float* g2 = (const float*)d_in[23]; const float* be2 = (const float*)d_in[24];
    const float* g3 = (const float*)d_in[25]; const float* be3 = (const float*)d_in[26];

    // workspace layout (halfs for transposed f16 weights, then f32 + f16 LN'd h_V)
    _Float16* base = (_Float16*)d_ws;
    _Float16* W1t   = base;
    _Float16* W2t   = W1t  + 128 * 384;
    _Float16* W3t   = W2t  + 128 * 128;
    _Float16* W11t  = W3t  + 128 * 128;
    _Float16* W12t  = W11t + 128 * 384;
    _Float16* W13t  = W12t + 128 * 128;
    _Float16* Wint  = W13t + 128 * 128;   // [512][128]
    _Float16* Woutt = Wint + 512 * 128;   // [128][512]
    float*    hln_f32 = (float*)(Woutt + 128 * 512);
    _Float16* hln_h   = (_Float16*)(hln_f32 + (size_t)NNODES * H);

    auto cv = [&](const float* s, _Float16* d, int Kd, int Nd) {
        int n = Kd * Nd;
        convert_tr_kernel<<<(n + 255) / 256, 256, 0, stream>>>(s, d, Kd, Nd);
    };
    cv(W1, W1t, 384, 128);  cv(W2, W2t, 128, 128);  cv(W3, W3t, 128, 128);
    cv(W11, W11t, 384, 128); cv(W12, W12t, 128, 128); cv(W13, W13t, 128, 128);
    cv(Win, Wint, 128, 512); cv(Wout, Woutt, 512, 128);

    float* outV = (float*)d_out;
    float* outE = outV + (size_t)NNODES * H;

    node_msg_kernel<<<NNODES, 128, 0, stream>>>(hV, hE, Eidx, maskA,
                                                W1t, b1, W2t, b2, W3t, b3,
                                                g1, be1, hln_f32, hln_h);
    ffn_kernel<<<NNODES / 16, 128, 0, stream>>>(hln_h, hln_f32, Wint, binp, Woutt, boutp,
                                                g2, be2, maskV, outV);
    edge_kernel<<<NNODES, 128, 0, stream>>>(outV, hE, Eidx,
                                            W11t, b11, W12t, b12, W13t, b13,
                                            g3, be3, outE);
}
